// GPT2_25305947308629
// MI455X (gfx1250) — matrix operand order repaired and checked by reading, not compiled, not run on hardware
//
#include <hip/hip_runtime.h>
#include <math.h>

// ---------------------------------------------------------------------------
// GPT-2 small forward on gfx1250 (wave32, WMMA bf16 16x16x32, big-LDS attn).
// ---------------------------------------------------------------------------

typedef __attribute__((ext_vector_type(16))) __bf16 v16bf;
typedef __attribute__((ext_vector_type(8)))  float  v8f;

union FragAB { v16bf v; unsigned short u[16]; };

__device__ __forceinline__ unsigned short f32_to_bf16(float f) {
  union { float f; unsigned int u; } a; a.f = f;
  unsigned int r = (a.u + 0x7FFFu + ((a.u >> 16) & 1u)) >> 16;
  return (unsigned short)r;
}
__device__ __forceinline__ float bf16_to_f32(unsigned short u) {
  union { unsigned int u; float f; } a; a.u = ((unsigned int)u) << 16;
  return a.f;
}

// ---------------------------------------------------------------------------
// Register-blocked GEMM: C[M,N] = act(A[M,K] * B[K,N] + bias) + residual
// Block: 256 threads = 8 waves; block tile 64(M) x 128(N); k-step 32.
// Wave grid 2(M) x 4(N); each wave owns a 32x32 output = 2x2 WMMA tiles
// (2 A-frags x 2 B-frags -> 4 v_wmma per k-step, amortizing staging+barrier).
// A/B staged fp32->bf16 in LDS (padded vs bank conflicts); next k-tile
// prefetched via global_prefetch_b8. act: 0 = none, 1 = exact GELU.
// ---------------------------------------------------------------------------
#define GEMM_BM 64
#define GEMM_BN 128
#define GEMM_BK 32

__global__ __launch_bounds__(256) void gemm_bf16_wmma(
    const float* __restrict__ A, const float* __restrict__ B,
    const float* __restrict__ bias, const float* __restrict__ residual,
    float* __restrict__ C, int M, int N, int K, int act)
{
  __shared__ unsigned short sA[GEMM_BM][GEMM_BK + 8];   // 64x40 -> 5 KB
  __shared__ unsigned short sB[GEMM_BK][GEMM_BN + 8];   // 32x136 -> 8.5 KB

  const int tid   = threadIdx.x;
  const int lane  = tid & 31;
  const int wave  = tid >> 5;
  const int wm    = wave >> 2;   // 0..1
  const int wn    = wave & 3;    // 0..3
  const int m0    = blockIdx.y * GEMM_BM;
  const int n0    = blockIdx.x * GEMM_BN;
  const int r     = lane & 15;
  const int khalf = lane >> 4;

  v8f acc[2][2] = {};

  for (int k0 = 0; k0 < K; k0 += GEMM_BK) {
    // Stage A tile (64x32): 2048 elements, 8 per thread, coalesced on k.
    #pragma unroll
    for (int i = 0; i < 8; ++i) {
      int idx = tid * 8 + i;
      int row = idx >> 5, col = idx & 31;
      sA[row][col] = f32_to_bf16(A[(long)(m0 + row) * K + (k0 + col)]);
    }
    // Stage B tile (32x128): 4096 elements, 16 per thread, coalesced on n.
    #pragma unroll
    for (int i = 0; i < 16; ++i) {
      int idx = tid * 16 + i;
      int row = idx >> 7, col = idx & 127;
      sB[row][col] = f32_to_bf16(B[(long)(k0 + row) * N + (n0 + col)]);
    }
    // Prefetch next k-tile while this one is consumed (global_prefetch_b8).
    if (k0 + GEMM_BK < K) {
      __builtin_prefetch(A + (long)(m0 + (tid & 63)) * K + (k0 + GEMM_BK), 0, 1);
      __builtin_prefetch(B + (long)(k0 + GEMM_BK + (tid & 31)) * N + n0 + (tid >> 1), 0, 1);
    }
    __syncthreads();

    // A fragments: lane holds row, K pattern per ISA 16-bit A table.
    FragAB af[2], bf[2];
    #pragma unroll
    for (int s = 0; s < 2; ++s) {
      int rowA = wm * 32 + s * 16 + r;
      #pragma unroll
      for (int v = 0; v < 8; ++v) {
        int kb = ((v < 4) ? 0 : 16) + khalf * 8 + (v & 3) * 2;
        af[s].u[2 * v]     = sA[rowA][kb];
        af[s].u[2 * v + 1] = sA[rowA][kb + 1];
      }
      // B fragments: lane = N column, 16 halves = K rows in table order.
      #pragma unroll
      for (int j = 0; j < 16; ++j)
        bf[s].u[j] = sB[((j < 8) ? 0 : 16) + khalf * 8 + (j & 7)][wn * 32 + s * 16 + r];
    }

    #pragma unroll
    for (int i = 0; i < 2; ++i)
      #pragma unroll
      for (int j = 0; j < 2; ++j)
        acc[i][j] = __builtin_amdgcn_wmma_f32_16x16x32_bf16(
            false, af[i].v, false, bf[j].v, (short)0, acc[i][j], false, false);
    __syncthreads();
  }

  // Epilogue: bias + optional GELU + optional residual, fp32 store.
  #pragma unroll
  for (int i = 0; i < 2; ++i)
    #pragma unroll
    for (int j = 0; j < 2; ++j)
      #pragma unroll
      for (int v = 0; v < 8; ++v) {
        int m = m0 + wm * 32 + i * 16 + v + khalf * 8;
        int n = n0 + wn * 32 + j * 16 + r;
        float val = acc[i][j][v] + (bias ? bias[n] : 0.0f);
        if (act == 1) val = 0.5f * val * (1.0f + erff(val * 0.7071067811865475f));
        if (residual) val += residual[(long)m * N + n];
        C[(long)m * N + n] = val;
      }
}

// ---------------------------------------------------------------------------
// Fused causal attention for one (b, h, 16-query tile): QK^T via WMMA into a
// 16x1024 *bf16* score strip in LDS (33 KB -> good WGP occupancy), row
// softmax in f32, unnormalized exp kept in bf16, then E*V via WMMA with the
// 1/rowsum normalization folded into the accumulator epilogue (exact: rinv is
// constant per output row). P fragments are raw ushort LDS copies -> pure
// ds_load_b128, zero VALU conversion in the PV inner loop.
// qkv layout: [B, S, 3H]; q at col h*64, k at 768+h*64, v at 1536+h*64.
// ---------------------------------------------------------------------------
__global__ __launch_bounds__(32) void attention_wmma(
    const float* __restrict__ qkv, float* __restrict__ attn_out, int Sdim)
{
  __shared__ unsigned short sE[16][1032];  // 16 x (1024+8 pad) bf16 ~ 33 KB
  __shared__ float sRinv[16];

  const int qt    = blockIdx.x;   // S/16 query tiles
  const int h     = blockIdx.y;   // head
  const int b     = blockIdx.z;   // batch
  const int lane  = threadIdx.x;
  const int r     = lane & 15;
  const int khalf = lane >> 4;
  const int H3    = 3 * 768;
  const float qscale = 0.125f;    // 1/sqrt(64), folded into Q

  const float* base = qkv + (long)b * Sdim * H3;

  // Q fragments for d=[0,32) and [32,64), pre-scaled, bf16.
  FragAB aq[2];
  #pragma unroll
  for (int f = 0; f < 2; ++f) {
    const float* qrow = base + (long)(qt * 16 + r) * H3 + h * 64;
    #pragma unroll
    for (int v = 0; v < 8; ++v) {
      int kb = ((v < 4) ? 0 : 16) + khalf * 8 + (v & 3) * 2;
      aq[f].u[2 * v]     = f32_to_bf16(qrow[f * 32 + kb] * qscale);
      aq[f].u[2 * v + 1] = f32_to_bf16(qrow[f * 32 + kb + 1] * qscale);
    }
  }

  // ---- scores = Q K^T, masked, into bf16 LDS strip ----
  for (int kt = 0; kt < Sdim / 16; ++kt) {
    v8f c = {};
    #pragma unroll
    for (int f = 0; f < 2; ++f) {
      FragAB bk;  // B[d][n] = K[kt*16+n][f*32+d], n = lane%16, d in table order
      #pragma unroll
      for (int j = 0; j < 16; ++j)
        bk.u[j] = f32_to_bf16(
            base[(long)(kt * 16 + r) * H3 + 768 + h * 64 + f * 32 + ((j < 8) ? 0 : 16) + khalf * 8 + (j & 7)]);
      c = __builtin_amdgcn_wmma_f32_16x16x32_bf16(false, aq[f].v, false, bk.v,
                                                  (short)0, c, false, false);
    }
    #pragma unroll
    for (int v = 0; v < 8; ++v) {
      int m = v + khalf * 8;
      int n = r;
      int qi = qt * 16 + m, ki = kt * 16 + n;
      float val = c[v];
      // faithful to ref: tril then masked_fill(==0, -1e4)
      if (ki > qi || val == 0.0f) val = -10000.0f;
      sE[m][kt * 16 + n] = f32_to_bf16(val);
    }
  }
  __syncthreads();

  // ---- row softmax over 1024 entries (16 rows, one lane each); keep the
  //      exp strip UNnormalized in bf16, remember 1/sum per row. ----
  if (lane < 16) {
    float mx = -3.0e38f;
    for (int j = 0; j < Sdim; ++j) mx = fmaxf(mx, bf16_to_f32(sE[lane][j]));
    float s = 0.0f;
    for (int j = 0; j < Sdim; ++j) {
      float e = expf(bf16_to_f32(sE[lane][j]) - mx);
      sE[lane][j] = f32_to_bf16(e);
      s += e;
    }
    sRinv[lane] = 1.0f / s;
  }
  __syncthreads();

  // ---- out = rinv * (E * V) ----
  v8f o[4] = {};
  for (int kst = 0; kst < Sdim / 32; ++kst) {
    FragAB ap;  // raw bf16 copy from LDS: vectorizes to ds_load_b128
    #pragma unroll
    for (int v = 0; v < 8; ++v) {
      int kb = ((v < 4) ? 0 : 16) + khalf * 8 + (v & 3) * 2;
      int kk = kst * 32 + kb;
      ap.u[2 * v]     = sE[r][kk];
      ap.u[2 * v + 1] = sE[r][kk + 1];
    }
    #pragma unroll
    for (int dt = 0; dt < 4; ++dt) {
      FragAB bv;  // B[k][n] = V[kst*32+k][dt*16+n], n = lane%16, k in table order
      #pragma unroll
      for (int j = 0; j < 16; ++j)
        bv.u[j] = f32_to_bf16(
            base[(long)(kst * 32 + ((j < 8) ? 0 : 16) + khalf * 8 + (j & 7)) * H3 + 1536 + h * 64 + dt * 16 + r]);
      o[dt] = __builtin_amdgcn_wmma_f32_16x16x32_bf16(false, ap.v, false, bv.v,
                                                      (short)0, o[dt], false, false);
    }
  }

  #pragma unroll
  for (int dt = 0; dt < 4; ++dt)
    #pragma unroll
    for (int v = 0; v < 8; ++v) {
      int m = v + khalf * 8;
      float rv = sRinv[m];
      attn_out[((long)b * Sdim + qt * 16 + m) * 768 + h * 64 + dt * 16 + r] =
          o[dt][v] * rv;
    }
}

// ---------------------------------------------------------------------------
// Row LayerNorm: one 256-thread block per row.
// ---------------------------------------------------------------------------
__global__ __launch_bounds__(256) void layernorm_rows(
    const float* __restrict__ X, const float* __restrict__ g,
    const float* __restrict__ bta, float* __restrict__ Y, int Hdim)
{
  __shared__ float red[256];
  const int tid = threadIdx.x;
  const float* x = X + (long)blockIdx.x * Hdim;
  float* y = Y + (long)blockIdx.x * Hdim;

  float s = 0.0f;
  for (int i = tid; i < Hdim; i += 256) s += x[i];
  red[tid] = s; __syncthreads();
  for (int o = 128; o > 0; o >>= 1) { if (tid < o) red[tid] += red[tid + o]; __syncthreads(); }
  float mu = red[0] / Hdim;
  __syncthreads();

  float v = 0.0f;
  for (int i = tid; i < Hdim; i += 256) { float d = x[i] - mu; v += d * d; }
  red[tid] = v; __syncthreads();
  for (int o = 128; o > 0; o >>= 1) { if (tid < o) red[tid] += red[tid + o]; __syncthreads(); }
  float rstd = rsqrtf(red[0] / Hdim + 1e-5f);

  for (int i = tid; i < Hdim; i += 256) y[i] = (x[i] - mu) * rstd * g[i] + bta[i];
}

// Final LN on last-token rows only -> enc[2,H] (written straight into d_out).
__global__ __launch_bounds__(256) void final_ln_lastrow(
    const float* __restrict__ X, const float* __restrict__ g,
    const float* __restrict__ bta, float* __restrict__ enc, int Sdim, int Hdim)
{
  __shared__ float red[256];
  const int tid = threadIdx.x;
  const float* x = X + ((long)blockIdx.x * Sdim + (Sdim - 1)) * Hdim;
  float* y = enc + (long)blockIdx.x * Hdim;

  float s = 0.0f;
  for (int i = tid; i < Hdim; i += 256) s += x[i];
  red[tid] = s; __syncthreads();
  for (int o = 128; o > 0; o >>= 1) { if (tid < o) red[tid] += red[tid + o]; __syncthreads(); }
  float mu = red[0] / Hdim;
  __syncthreads();

  float v = 0.0f;
  for (int i = tid; i < Hdim; i += 256) { float d = x[i] - mu; v += d * d; }
  red[tid] = v; __syncthreads();
  for (int o = 128; o > 0; o >>= 1) { if (tid < o) red[tid] += red[tid + o]; __syncthreads(); }
  float rstd = rsqrtf(red[0] / Hdim + 1e-5f);

  for (int i = tid; i < Hdim; i += 256) y[i] = (x[i] - mu) * rstd * g[i] + bta[i];
}

// x = wte[ids] + wpe
__global__ __launch_bounds__(256) void embed_kernel(
    const int* __restrict__ ids, const float* __restrict__ wte,
    const float* __restrict__ wpe, float* __restrict__ X, int Sdim, int Hdim)
{
  const int row = blockIdx.x;          // b*S + s
  const int s   = row % Sdim;
  const int tok = ids[row];
  const float* e = wte + (long)tok * Hdim;
  const float* p = wpe + (long)s * Hdim;
  float* x = X + (long)row * Hdim;
  for (int i = threadIdx.x; i < Hdim; i += 256) x[i] = e[i] + p[i];
}

// Tied unembed GEMV: logits[b][v] = enc[b] . wte[v]. Memory-bound (154MB wte
// stream at 23.3 TB/s ~ 7us); plain VALU dot is the right tool here.
__global__ __launch_bounds__(256) void unembed_kernel(
    const float* __restrict__ enc, const float* __restrict__ wte,
    float* __restrict__ logits, int Vdim, int Hdim)
{
  int v = blockIdx.x * 256 + threadIdx.x;
  if (v >= Vdim) return;
  const float* w = wte + (long)v * Hdim;
  float s0 = 0.0f, s1 = 0.0f;
  for (int i = 0; i < Hdim; ++i) {
    float wv = w[i];
    s0 += enc[i] * wv;
    s1 += enc[Hdim + i] * wv;
  }
  logits[v] = s0;
  logits[(long)Vdim + v] = s1;
}

// ---------------------------------------------------------------------------
// Host orchestration.
// ---------------------------------------------------------------------------
extern "C" void kernel_launch(void* const* d_in, const int* in_sizes, int n_in,
                              void* d_out, int out_size, void* d_ws, size_t ws_size,
                              hipStream_t stream) {
  (void)in_sizes; (void)n_in; (void)out_size; (void)ws_size;

  const int L = 12, H = 768, NH = 12, V = 50257, S = 1024, B = 2;
  const int M = B * S;  // 2048 token rows

  const int*   ids   = (const int*)  d_in[0];
  const float* wte   = (const float*)d_in[1];
  const float* wpe   = (const float*)d_in[2];
  const float* ln1_g = (const float*)d_in[3];
  const float* ln1_b = (const float*)d_in[4];
  const float* Wqkv  = (const float*)d_in[5];
  const float* bqkv  = (const float*)d_in[6];
  const float* Wo    = (const float*)d_in[7];
  const float* bo    = (const float*)d_in[8];
  const float* ln2_g = (const float*)d_in[9];
  const float* ln2_b = (const float*)d_in[10];
  const float* W1    = (const float*)d_in[11];
  const float* b1    = (const float*)d_in[12];
  const float* W2    = (const float*)d_in[13];
  const float* b2    = (const float*)d_in[14];
  const float* lnf_g = (const float*)d_in[15];
  const float* lnf_b = (const float*)d_in[16];

  // Workspace carve-up (fp32): x, h, qkv, attn, mid  -> ~63 MB total.
  float* x    = (float*)d_ws;
  float* h    = x    + (size_t)M * H;
  float* qkvb = h    + (size_t)M * H;
  float* attn = qkvb + (size_t)M * 3 * H;
  float* mid  = attn + (size_t)M * H;

  float* logits = (float*)d_out;          // [2, V]
  float* enc    = logits + 2 * (size_t)V; // [2, H] (second return value)

  embed_kernel<<<M, 256, 0, stream>>>(ids, wte, wpe, x, S, H);

  for (int l = 0; l < L; ++l) {
    layernorm_rows<<<M, 256, 0, stream>>>(x, ln1_g + (size_t)l * H,
                                          ln1_b + (size_t)l * H, h, H);
    gemm_bf16_wmma<<<dim3(3 * H / GEMM_BN, M / GEMM_BM), 256, 0, stream>>>(
        h, Wqkv + (size_t)l * H * 3 * H, bqkv + (size_t)l * 3 * H, nullptr,
        qkvb, M, 3 * H, H, 0);
    attention_wmma<<<dim3(S / 16, NH, B), 32, 0, stream>>>(qkvb, attn, S);
    gemm_bf16_wmma<<<dim3(H / GEMM_BN, M / GEMM_BM), 256, 0, stream>>>(
        attn, Wo + (size_t)l * H * H, bo + (size_t)l * H, x, x, M, H, H, 0);
    layernorm_rows<<<M, 256, 0, stream>>>(x, ln2_g + (size_t)l * H,
                                          ln2_b + (size_t)l * H, h, H);
    gemm_bf16_wmma<<<dim3(4 * H / GEMM_BN, M / GEMM_BM), 256, 0, stream>>>(
        h, W1 + (size_t)l * H * 4 * H, b1 + (size_t)l * 4 * H, nullptr,
        mid, M, 4 * H, H, 1);
    gemm_bf16_wmma<<<dim3(H / GEMM_BN, M / GEMM_BM), 256, 0, stream>>>(
        mid, W2 + (size_t)l * 4 * H * H, b2 + (size_t)l * H, x, x, M, H,
        4 * H, 0);
  }

  final_ln_lastrow<<<B, 256, 0, stream>>>(x, lnf_g, lnf_b, enc, S, H);
  unembed_kernel<<<(V + 255) / 256, 256, 0, stream>>>(enc, wte, logits, V, H);
}
